// MultiHeadAttentionWithRoPE_9131100471604
// MI455X (gfx1250) — matrix operand order repaired and checked by reading, not compiled, not run on hardware
//
#include <hip/hip_runtime.h>
#include <hip/hip_bf16.h>
#include <stdint.h>

// ---------------------------------------------------------------------------
// MultiHeadAttention forward for MI455X (gfx1250, wave32, WMMA).
// S=512, B=16, D=2048, H=4, hd=512. Compute-bound (~310 GFLOP vs ~0.5GB HBM)
// => f16 WMMA (v_wmma_f32_16x16x32_f16) with f32 accumulation.
// LDS tiles are stored in *fragment-major* order (frag[subtile][lane][16]),
// so every WMMA operand load is a contiguous 32B ds_load_b128 pair with a
// wave-uniform base — no per-lane select math, no sub-dword stitching.
// ---------------------------------------------------------------------------

typedef __attribute__((ext_vector_type(16))) _Float16 v16h;
typedef __attribute__((ext_vector_type(8)))  float    v8f;

static __device__ inline v8f wmma32f16(v16h a, v16h b, v8f c) {
  // emits v_wmma_f32_16x16x32_f16 (8-arg form, probe-confirmed)
  return __builtin_amdgcn_wmma_f32_16x16x32_f16(false, a, false, b, (short)0, c,
                                                false, false);
}

static __device__ inline v8f zero8() {
  v8f z;
#pragma unroll
  for (int j = 0; j < 8; ++j) z[j] = 0.0f;
  return z;
}

// Contiguous 32B fragment load from LDS -> 2x ds_load_b128.
static __device__ inline v16h load_frag32(const _Float16* p) {
  union { uint4 u[2]; v16h v; } t;
  t.u[0] = *(const uint4*)p;
  t.u[1] = *(const uint4*)(p + 8);
  return t.v;
}

// 8x f32 -> 8x f16 packed into a uint4 (for b128 LDS stores).
static __device__ inline uint4 cvt8(const float4 a, const float4 b) {
  union { _Float16 h[8]; uint4 u; } t;
  t.h[0] = (_Float16)a.x; t.h[1] = (_Float16)a.y;
  t.h[2] = (_Float16)a.z; t.h[3] = (_Float16)a.w;
  t.h[4] = (_Float16)b.x; t.h[5] = (_Float16)b.y;
  t.h[6] = (_Float16)b.z; t.h[7] = (_Float16)b.w;
  return t.u;
}

// Fragment-major placement (this file's contiguous k order, same for A and B):
//  A 16x32 f16: lane = M + 16*(K>=16);    elem = K&15
//  B 32x16 f16: lane = N + 16*(K>=16);    elem = K&15
// Staging threads write these layouts with aligned uint4 stores.

// ---------------------------------------------------------------------------
// Kernel 1: projection GEMM.  C[8192,2048] = X[8192,2048] @ W[2048,2048]^T,
// f32 in, f16 out scattered to head-major (H*B, S, hd); VT=true writes the
// transposed (H*B, hd, S) layout (used for v so attn@v's B tile is K-rowed).
// Grid (16,64): 128x128 tile per WG; 8 waves, each 16 rows x 128 cols.
// ---------------------------------------------------------------------------
template <bool VT>
__global__ __launch_bounds__(256) void proj_gemm_kernel(
    const float* __restrict__ X, const float* __restrict__ W,
    _Float16* __restrict__ dst) {
  __shared__ __align__(32) _Float16 Af[8][32][16];  // 8KB
  __shared__ __align__(32) _Float16 Bf[8][32][16];  // 8KB
  const int tid = threadIdx.x;
  const int lane = tid & 31, wave = tid >> 5;
  const int m0 = blockIdx.y << 7, n0 = blockIdx.x << 7;
  const int lrow = tid >> 1;            // 0..127
  const int lseg = (tid & 1) << 4;      // 0 or 16 (K offset, floats)
  const int eb = lseg ? 8 : 0;          // unused; A also uses lane bl, elems 0..15
  const int sub = lrow >> 4, r = lrow & 15;
  const int bl = r + (lseg ? 16 : 0);   // B lane for this K half

  v8f acc[8];
#pragma unroll
  for (int t = 0; t < 8; ++t) acc[t] = zero8();

  const float* agp = X + (size_t)(m0 + lrow) * 2048 + lseg;
  const float* bgp = W + (size_t)(n0 + lrow) * 2048 + lseg;

  for (int kk = 0; kk < 2048; kk += 32) {
    __builtin_prefetch(agp + kk + 64, 0, 1);  // global_prefetch_b8, next panel
    __builtin_prefetch(bgp + kk + 64, 0, 1);
    const float4* a4 = (const float4*)(agp + kk);
    const float4* b4 = (const float4*)(bgp + kk);
    // A: 16 K-contiguous halfs of row m -> lane bl, elems 0..15
    *(uint4*)&Af[sub][bl][0] = cvt8(a4[0], a4[1]);
    *(uint4*)&Af[sub][bl][8] = cvt8(a4[2], a4[3]);
    // B: 16 K-contiguous halfs of column n -> one lane, elems 0..15
    *(uint4*)&Bf[sub][bl][0] = cvt8(b4[0], b4[1]);
    *(uint4*)&Bf[sub][bl][8] = cvt8(b4[2], b4[3]);
    __syncthreads();
    v16h af = load_frag32(&Af[wave][lane][0]);
#pragma unroll
    for (int t = 0; t < 8; ++t) {
      v16h bf = load_frag32(&Bf[t][lane][0]);
      acc[t] = wmma32f16(af, bf, acc[t]);
    }
    __syncthreads();
  }

  // C/D layout: VGPR i -> row (lane<16 ? i : i+8), col = lane%16 within tile.
#pragma unroll
  for (int t = 0; t < 8; ++t) {
    const int col = n0 + (t << 4) + (lane & 15);
    const int h = col >> 9, d = col & 511;
#pragma unroll
    for (int i = 0; i < 8; ++i) {
      const int m = m0 + (wave << 4) + ((lane < 16) ? i : i + 8);
      const int s = m >> 4, bb = m & 15;
      if (VT)
        dst[(((size_t)(h * 16 + bb) * 512 + d) << 9) + s] = (_Float16)acc[t][i];
      else
        dst[(((size_t)(h * 16 + bb) * 512 + s) << 9) + d] = (_Float16)acc[t][i];
    }
  }
}

// ---------------------------------------------------------------------------
// Kernel 2: scores = q@k^T / sqrt(hd), row softmax over all 512 keys,
// write attn f32 (attn_score region of d_out) + attn f16 (workspace).
// Grid (32,64): WG = one batch n, 16 query rows; 8 waves x 64 key cols.
// Dynamic LDS 67,648 B: Qf[32][16]h | Kf[32][32][16]h | Ss[16*512]f | red
// ---------------------------------------------------------------------------
__global__ __launch_bounds__(256) void scores_softmax_kernel(
    const _Float16* __restrict__ qf, const _Float16* __restrict__ kf,
    float* __restrict__ attn_out, _Float16* __restrict__ attn16) {
  extern __shared__ __align__(32) char smem_raw[];
  _Float16* Qf = (_Float16*)smem_raw;               // 32*16 halfs   (1024 B)
  _Float16* Kf = (_Float16*)(smem_raw + 1024);      // 32*32*16 halfs (32 KB)
  float* Ss = (float*)(smem_raw + 33792);           // 16*512 f32     (32 KB)
  float* red = (float*)(smem_raw + 66560);          // 16*17 f32

  const int tid = threadIdx.x, lane = tid & 31, wave = tid >> 5;
  const int n = blockIdx.y, s0 = blockIdx.x << 4;
  const int h = n >> 4, bb = n & 15;
  const _Float16* qbase = qf + (((size_t)n * 512 + s0) << 9);
  const _Float16* kbase = kf + ((size_t)n << 18);

  v8f acc[4];
#pragma unroll
  for (int t = 0; t < 4; ++t) acc[t] = zero8();

  for (int kk = 0; kk < 512; kk += 32) {
    if (tid < 64) {  // Qf: 16 rows x 32 halfs, one 8-half chunk per thread
      const int r = tid >> 2, c = (tid & 3) << 3;
      const int ln = r + (((c >> 4) & 1) ? 16 : 0);
      const int e0 = ((c >> 3) & 1) ? 8 : 0;
      *(uint4*)&Qf[ln * 16 + e0] =
          *(const uint4*)(qbase + ((size_t)r << 9) + kk + c);
    }
#pragma unroll
    for (int i = 0; i < 2; ++i) {  // Kf: 512 key rows, 2 rows / thread
      const int kr = (tid << 1) + i;
      const uint4* s4 = (const uint4*)(kbase + ((size_t)kr << 9) + kk);
      _Float16* lo = &Kf[(((kr >> 4) << 5) + (kr & 15)) << 4];
      _Float16* hi = lo + (16 << 4);
      *(uint4*)(lo) = s4[0];      // K 0-7   -> low lane elems 0-7
      *(uint4*)(lo + 8) = s4[1];  // K 8-15  -> low lane elems 8-15
      *(uint4*)(hi) = s4[2];      // K 16-23 -> high lane elems 0-7
      *(uint4*)(hi + 8) = s4[3];  // K 24-31 -> high lane elems 8-15
    }
    __syncthreads();
    v16h af = load_frag32(&Qf[lane << 4]);
#pragma unroll
    for (int t = 0; t < 4; ++t) {
      v16h bf = load_frag32(&Kf[((((wave << 2) + t) << 5) + lane) << 4]);
      acc[t] = wmma32f16(af, bf, acc[t]);
    }
    __syncthreads();
  }

  const float scale = 0.044194173824159216f;  // 1/sqrt(512)
#pragma unroll
  for (int t = 0; t < 4; ++t) {
    const int col = (wave << 6) + (t << 4) + (lane & 15);
#pragma unroll
    for (int i = 0; i < 8; ++i) {
      const int row = (lane < 16) ? i : i + 8;
      Ss[(row << 9) + col] = acc[t][i] * scale;
    }
  }
  __syncthreads();

  // Softmax: 16 threads per row, 32 columns per thread.
  const int row = tid >> 4, seg = tid & 15;
  float* sr = &Ss[row << 9];
  float mx = -3.0e38f;
#pragma unroll 4
  for (int c = 0; c < 32; ++c) mx = fmaxf(mx, sr[(seg << 5) + c]);
  red[row * 17 + seg] = mx;
  __syncthreads();
#pragma unroll
  for (int i = 0; i < 16; ++i) mx = fmaxf(mx, red[row * 17 + i]);
  float sum = 0.0f;
#pragma unroll 4
  for (int c = 0; c < 32; ++c) {
    const float e = __expf(sr[(seg << 5) + c] - mx);
    sr[(seg << 5) + c] = e;
    sum += e;
  }
  __syncthreads();
  red[row * 17 + seg] = sum;
  __syncthreads();
  sum = 0.0f;
#pragma unroll
  for (int i = 0; i < 16; ++i) sum += red[row * 17 + i];
  const float inv = 1.0f / sum;

  const int sg = s0 + row;
  float* ao = attn_out + ((((size_t)bb * 512 + sg) * 4 + h) << 9);  // (B,S,H,hd)
  _Float16* a16 = attn16 + (((size_t)n * 512 + sg) << 9);           // (64,S,S)
  for (int c = 0; c < 32; ++c) {
    const int col = (seg << 5) + c;
    const float p = sr[col] * inv;
    ao[col] = p;
    a16[col] = (_Float16)p;
  }
}

// ---------------------------------------------------------------------------
// Kernel 3: out = attn(512x512) @ v(512x512) per batch; both operands f16,
// v supplied transposed (n, d, t) so its tile stages with K-contiguous rows.
// Output f16 into the reference's flat-reinterpret layout ((b*H+h)*S+s)*hd+d.
// Grid (4,4,64): 128x128 tile per WG.
// ---------------------------------------------------------------------------
__global__ __launch_bounds__(256) void attn_v_kernel(
    const _Float16* __restrict__ attn16, const _Float16* __restrict__ vt,
    _Float16* __restrict__ ctx) {
  __shared__ __align__(32) _Float16 Af[8][32][16];
  __shared__ __align__(32) _Float16 Bf[8][32][16];
  const int tid = threadIdx.x, lane = tid & 31, wave = tid >> 5;
  const int n = blockIdx.z, s0 = blockIdx.y << 7, d0 = blockIdx.x << 7;
  const int h = n >> 4, bb = n & 15;
  const _Float16* abase = attn16 + ((size_t)n << 18);
  const _Float16* tbase = vt + ((size_t)n << 18);

  v8f acc[8];
#pragma unroll
  for (int t = 0; t < 8; ++t) acc[t] = zero8();

  const int lrow = tid >> 1, lseg = (tid & 1) << 4;
  const int eb = lseg ? 8 : 0;
  const int sub = lrow >> 4, r = lrow & 15;
  const int bl = r + (lseg ? 16 : 0);

  for (int kk = 0; kk < 512; kk += 32) {
    __builtin_prefetch(abase + ((size_t)(s0 + lrow) << 9) + kk + lseg + 64, 0, 1);
    {  // A tile: attn rows s0..s0+127, K=t
      const uint4* s4 =
          (const uint4*)(abase + ((size_t)(s0 + lrow) << 9) + kk + lseg);
      *(uint4*)&Af[sub][bl][0] = s4[0];
      *(uint4*)&Af[sub][bl][8] = s4[1];
    }
    {  // B tile: v^T rows d0..d0+127, K=t contiguous
      const uint4* s4 =
          (const uint4*)(tbase + ((size_t)(d0 + lrow) << 9) + kk + lseg);
      *(uint4*)&Bf[sub][bl][0] = s4[0];
      *(uint4*)&Bf[sub][bl][8] = s4[1];
    }
    __syncthreads();
    v16h af = load_frag32(&Af[wave][lane][0]);
#pragma unroll
    for (int t = 0; t < 8; ++t) {
      v16h bf = load_frag32(&Bf[t][lane][0]);
      acc[t] = wmma32f16(af, bf, acc[t]);
    }
    __syncthreads();
  }

#pragma unroll
  for (int t = 0; t < 8; ++t) {
    const int d = d0 + (t << 4) + (lane & 15);
#pragma unroll
    for (int i = 0; i < 8; ++i) {
      const int s = s0 + (wave << 4) + ((lane < 16) ? i : i + 8);
      ctx[(((size_t)(bb * 4 + h) * 512 + s) << 9) + d] = (_Float16)acc[t][i];
    }
  }
}

// ---------------------------------------------------------------------------
// Kernel 4: output = ctx(8192x2048 f16, flat view) @ Wo^T (f32->f16), f32 out.
// ---------------------------------------------------------------------------
__global__ __launch_bounds__(256) void out_proj_kernel(
    const _Float16* __restrict__ ctx, const float* __restrict__ Wo,
    float* __restrict__ out) {
  __shared__ __align__(32) _Float16 Af[8][32][16];
  __shared__ __align__(32) _Float16 Bf[8][32][16];
  const int tid = threadIdx.x, lane = tid & 31, wave = tid >> 5;
  const int m0 = blockIdx.y << 7, n0 = blockIdx.x << 7;
  const int lrow = tid >> 1, lseg = (tid & 1) << 4;
  const int eb = lseg ? 8 : 0;
  const int sub = lrow >> 4, r = lrow & 15;
  const int bl = r + (lseg ? 16 : 0);

  v8f acc[8];
#pragma unroll
  for (int t = 0; t < 8; ++t) acc[t] = zero8();

  const _Float16* agp = ctx + ((size_t)(m0 + lrow) << 11) + lseg;
  const float* bgp = Wo + ((size_t)(n0 + lrow) << 11) + lseg;

  for (int kk = 0; kk < 2048; kk += 32) {
    __builtin_prefetch(agp + kk + 64, 0, 1);
    __builtin_prefetch(bgp + kk + 64, 0, 1);
    {  // A (already f16): direct b128 copies into fragment layout
      const uint4* s4 = (const uint4*)(agp + kk);
      *(uint4*)&Af[sub][bl][0] = s4[0];
      *(uint4*)&Af[sub][bl][8] = s4[1];
    }
    {  // B (f32 weights): convert + pack
      const float4* b4 = (const float4*)(bgp + kk);
      *(uint4*)&Bf[sub][bl][0] = cvt8(b4[0], b4[1]);
      *(uint4*)&Bf[sub][bl][8] = cvt8(b4[2], b4[3]);
    }
    __syncthreads();
    v16h af = load_frag32(&Af[wave][lane][0]);
#pragma unroll
    for (int t = 0; t < 8; ++t) {
      v16h bf = load_frag32(&Bf[t][lane][0]);
      acc[t] = wmma32f16(af, bf, acc[t]);
    }
    __syncthreads();
  }

#pragma unroll
  for (int t = 0; t < 8; ++t) {
    const int col = n0 + (t << 4) + (lane & 15);
#pragma unroll
    for (int i = 0; i < 8; ++i) {
      const int m = m0 + (wave << 4) + ((lane < 16) ? i : i + 8);
      out[(size_t)m * 2048 + col] = acc[t][i];
    }
  }
}

// ---------------------------------------------------------------------------
// Launch. d_in: query, key, value, Wq, Wk, Wv, Wo (all f32).
// d_out: output (16,777,216 f32) ++ attn_score (16,777,216 f32).
// Workspace (128 MB): q16 | k16 | vT16 f16, attn f16; ctx aliases q16 (q is
// fully consumed by kernel 2 before kernel 3 writes; stream-ordered).
// ---------------------------------------------------------------------------
extern "C" void kernel_launch(void* const* d_in, const int* in_sizes, int n_in,
                              void* d_out, int out_size, void* d_ws,
                              size_t ws_size, hipStream_t stream) {
  const float* query = (const float*)d_in[0];
  const float* key   = (const float*)d_in[1];
  const float* value = (const float*)d_in[2];
  const float* Wq = (const float*)d_in[3];
  const float* Wk = (const float*)d_in[4];
  const float* Wv = (const float*)d_in[5];
  const float* Wo = (const float*)d_in[6];
  float* out = (float*)d_out;
  float* attn_out = out + (size_t)16777216;

  char* ws = (char*)d_ws;
  _Float16* qf  = (_Float16*)(ws);
  _Float16* kf  = (_Float16*)(ws + (size_t)33554432);
  _Float16* vtf = (_Float16*)(ws + (size_t)67108864);
  _Float16* a16 = (_Float16*)(ws + (size_t)100663296);
  _Float16* ctx = qf;

  const dim3 blk(256);
  proj_gemm_kernel<false><<<dim3(16, 64), blk, 0, stream>>>(query, Wq, qf);
  proj_gemm_kernel<false><<<dim3(16, 64), blk, 0, stream>>>(key, Wk, kf);
  proj_gemm_kernel<true><<<dim3(16, 64), blk, 0, stream>>>(value, Wv, vtf);
  scores_softmax_kernel<<<dim3(32, 64), blk, 67648, stream>>>(qf, kf, attn_out,
                                                              a16);
  attn_v_kernel<<<dim3(4, 4, 64), blk, 0, stream>>>(a16, vtf, ctx);
  out_proj_kernel<<<dim3(16, 64), blk, 0, stream>>>(ctx, Wo, out);

  (void)in_sizes; (void)n_in; (void)out_size; (void)ws_size;
}
